// RNN_73710228734670
// MI455X (gfx1250) — matrix operand order repaired and checked by reading, not compiled, not run on hardware
//
#include <hip/hip_runtime.h>
#include <hip/hip_bf16.h>
#include <math.h>

// ---------------------------------------------------------------------------
// Problem constants (from the reference)
// ---------------------------------------------------------------------------
constexpr int BB  = 1024;   // batch
constexpr int PP  = 3;      // P
constexpr int KK  = 64;     // K (top-k)
constexpr int EE  = 8000;   // vocabulary
constexpr int DE  = 64;
constexpr int DD  = 32;
constexpr int HH  = 256;

constexpr int FEAT_D   = KK * PP * (DE + DD);      // 18432
constexpr int LOGITS_D = HH * KK;                  // 16384
constexpr int RANKIN_D = LOGITS_D + 2 * KK * DE;   // 24576
constexpr int MLPIN_D  = DE + 2 * DD + HH;         // 384
constexpr int PRED_D   = EE * PP;                  // 24000
constexpr long OUT_PRED_OFF = 1L + (long)BB * PP * KK;  // loss + sorted preds

typedef __attribute__((ext_vector_type(16))) __bf16 bfx16;
typedef __attribute__((ext_vector_type(8)))  __bf16 bfx8;
typedef __attribute__((ext_vector_type(8)))  float  fx8;

// ---------------------------------------------------------------------------
// Generic WMMA bf16 GEMM:  C(MxN) = A(MxKd) * W(KdxN) + bias, opt ReLU.
// A: bf16 row-major, leading dim lda.  BT: bf16 = W^T, (N x Kd) row-major.
// Writes fp32 to Cf (ldc) and/or bf16 to Cb (ldo).  One wave per 16x64 tile.
// Requires: M%16==0, N%64==0, Kd%32==0, 16B-aligned rows (guaranteed here).
// ---------------------------------------------------------------------------
__global__ __launch_bounds__(256) void gemm_wmma_bf16(
    const __bf16* __restrict__ A, int lda,
    const __bf16* __restrict__ BT,
    const float*  __restrict__ bias,
    float*        __restrict__ Cf, int ldc,
    __bf16*       __restrict__ Cb, int ldo,
    int M, int N, int Kd, int relu)
{
  const int lane   = threadIdx.x & 31;
  const int wave   = blockIdx.x * (blockDim.x >> 5) + (threadIdx.x >> 5);
  const int nQuads = N >> 6;
  const int mTiles = M >> 4;
  if (wave >= mTiles * nQuads) return;          // wave-uniform: EXEC stays all-1s
  const int mBase  = (wave / nQuads) << 4;
  const int nBase  = (wave % nQuads) << 6;
  const int lane16 = lane & 15;
  const int hsel   = lane >> 4;                 // half-wave select

  const __bf16* aRow = A  + (size_t)(mBase + lane16) * lda;
  const __bf16* bRow = BT + (size_t)(nBase + lane16) * Kd;

  fx8 acc[4] = {fx8{0.f}, fx8{0.f}, fx8{0.f}, fx8{0.f}};

  for (int k0 = 0; k0 < Kd; k0 += 32) {
    // A k order chosen here: 16 contiguous K per half-wave (matches B).
    union { bfx16 v; bfx8 h[2]; } af;
    af.h[0] = *(const bfx8*)(aRow + k0 +     hsel * 16);
    af.h[1] = *(const bfx8*)(aRow + k0 + 8 + hsel * 16);
#pragma unroll
    for (int q = 0; q < 4; ++q) {
      // B k order chosen here: lane = column, 16 contiguous K per half-wave.
      bfx16 bf = *(const bfx16*)(bRow + (size_t)(q * 16) * Kd + k0 + hsel * 16);
      acc[q] = __builtin_amdgcn_wmma_f32_16x16x32_bf16(
          false, af.v, false, bf, (short)0, acc[q], false, false);
    }
  }

#pragma unroll
  for (int q = 0; q < 4; ++q) {
    const int col = nBase + q * 16 + lane16;
    const float bv = bias ? bias[col] : 0.0f;
#pragma unroll
    for (int r = 0; r < 8; ++r) {
      const int row = mBase + (hsel << 3) + r;   // C/D layout: VGPR r -> M=r / M=8+r
      float v = acc[q][r] + bv;
      if (relu) v = fmaxf(v, 0.0f);
      if (Cf) Cf[(size_t)row * ldc + col] = v;
      if (Cb) Cb[(size_t)row * ldo + col] = (__bf16)v;
    }
  }
}

// ---------------------------------------------------------------------------
// fp32 (Kd x N) -> bf16 transposed (N x Kd)
// ---------------------------------------------------------------------------
__global__ void k_transpose(const float* __restrict__ W, __bf16* __restrict__ WT,
                            int Kd, int N)
{
  size_t i = (size_t)blockIdx.x * blockDim.x + threadIdx.x;
  if (i >= (size_t)Kd * N) return;
  int n = (int)(i % N), k = (int)(i / N);
  WT[(size_t)n * Kd + k] = (__bf16)W[i];
}

// ---------------------------------------------------------------------------
// feat[b, (k*P+p)*96 + d] = d<64 ? link_emb[pred+1][d] : dir_emb[dir+1][d-64]
// ---------------------------------------------------------------------------
__global__ void k_feat(const int* __restrict__ preds, const int* __restrict__ dirlab,
                       const float* __restrict__ link_emb, const float* __restrict__ dir_emb,
                       __bf16* __restrict__ featB)
{
  size_t i = (size_t)blockIdx.x * blockDim.x + threadIdx.x;
  if (i >= (size_t)BB * FEAT_D) return;
  int d = (int)(i % 96);
  size_t t = i / 96;
  int p = (int)(t % 3); t /= 3;
  int k = (int)(t % 64);
  int b = (int)(t / 64);
  int e = preds[((size_t)b * 3 + p) * 64 + k];
  float v = (d < DE) ? link_emb[(size_t)(e + 1) * DE + d]
                     : dir_emb[(size_t)(dirlab[e] + 1) * DD + (d - DE)];
  featB[i] = (__bf16)v;
}

// S[b,p,d] = sum_k link_emb[pred[b,p,k]+1][d] * hyp[d]   (for conn and cons hyps)
__global__ void k_sums(const int* __restrict__ preds, const float* __restrict__ link_emb,
                       const float* __restrict__ ch, const float* __restrict__ sh,
                       float* __restrict__ Sconn, float* __restrict__ Scons)
{
  int i = blockIdx.x * blockDim.x + threadIdx.x;
  if (i >= BB * PP * DE) return;
  int d = i & 63, p = (i >> 6) % 3, b = i / (PP * DE);
  float chd = ch[d], shd = sh[d], sc = 0.f, ss = 0.f;
  const int* pr = preds + ((size_t)b * 3 + p) * 64;
  for (int k = 0; k < 64; ++k) {
    float pe = link_emb[(size_t)(pr[k] + 1) * DE + d];
    sc += pe * chd; ss += pe * shd;
  }
  Sconn[i] = sc; Scons[i] = ss;
}

// Telescoped means: m[b,k,d] = emb[d] + 0.5*((pe0-pe2) - hyp[d]*(S0-S2)).
// Written bf16 straight into rank_in columns [16384:20480]=cons, [20480:24576]=conn.
__global__ void k_connm(const int* __restrict__ preds, const float* __restrict__ link_emb,
                        const float* __restrict__ ch, const float* __restrict__ sh,
                        const float* __restrict__ conn_emb, const float* __restrict__ cons_emb,
                        const float* __restrict__ Sconn, const float* __restrict__ Scons,
                        __bf16* __restrict__ rank_in)
{
  int i = blockIdx.x * blockDim.x + threadIdx.x;
  if (i >= BB * KK * DE) return;
  int d = i & 63, k = (i >> 6) & 63, b = i >> 12;
  int e0 = preds[((size_t)b * 3 + 0) * 64 + k];
  int e2 = preds[((size_t)b * 3 + 2) * 64 + k];
  float dpe = link_emb[(size_t)(e0 + 1) * DE + d] - link_emb[(size_t)(e2 + 1) * DE + d];
  float dSc = Sconn[(b * 3 + 0) * 64 + d] - Sconn[(b * 3 + 2) * 64 + d];
  float dSs = Scons[(b * 3 + 0) * 64 + d] - Scons[(b * 3 + 2) * 64 + d];
  float connv = conn_emb[d] + 0.5f * (dpe - ch[d] * dSc);
  float consv = cons_emb[d] + 0.5f * (dpe - sh[d] * dSs);
  size_t base = (size_t)b * RANKIN_D;
  rank_in[base + LOGITS_D            + (size_t)k * 64 + d] = (__bf16)consv;
  rank_in[base + LOGITS_D + KK * DE  + (size_t)k * 64 + d] = (__bf16)connv;
}

__global__ void k_embs(const float* __restrict__ embs, __bf16* __restrict__ mlp_in)
{
  int i = blockIdx.x * blockDim.x + threadIdx.x;
  if (i >= BB * 128) return;
  int b = i >> 7, j = i & 127;
  mlp_in[(size_t)b * MLPIN_D + j] = (__bf16)embs[i];
}

__global__ void k_zero(float* accum) { if (threadIdx.x < 4) accum[threadIdx.x] = 0.f; }

// per (b,p): NLL of log_softmax over E=8000; accumulate into accum[0]
__global__ __launch_bounds__(256) void k_cls_loss(const float* __restrict__ pred,
                                                  const int* __restrict__ gt, float* accum)
{
  int b = blockIdx.x / 3, p = blockIdx.x % 3;
  const float* row = pred + (size_t)b * PRED_D + (size_t)p * EE;
  __shared__ float red[8];
  __shared__ float s_m, s_s;
  int tid = threadIdx.x;
  float m = -3.4e38f;
  for (int i = tid; i < EE; i += 256) m = fmaxf(m, row[i]);
  for (int o = 16; o > 0; o >>= 1) m = fmaxf(m, __shfl_xor(m, o, 32));
  if ((tid & 31) == 0) red[tid >> 5] = m;
  __syncthreads();
  if (tid == 0) { float mm = red[0]; for (int i = 1; i < 8; ++i) mm = fmaxf(mm, red[i]); s_m = mm; }
  __syncthreads();
  m = s_m;
  float s = 0.f;
  for (int i = tid; i < EE; i += 256) s += expf(row[i] - m);
  for (int o = 16; o > 0; o >>= 1) s += __shfl_xor(s, o, 32);
  if ((tid & 31) == 0) red[tid >> 5] = s;
  __syncthreads();
  if (tid == 0) {
    float ss = 0.f; for (int i = 0; i < 8; ++i) ss += red[i];
    float logZ = m + logf(ss);
    float nll = logZ - row[gt[b * 3 + p]];
    atomicAdd(&accum[0], nll);
  }
}

// per b: match-count argmax, log-softmax(64), argsort, masked CE, sorted preds out.
__global__ __launch_bounds__(64) void k_final(const float* __restrict__ rank_pred,
                                              const int* __restrict__ preds,
                                              const int* __restrict__ gt,
                                              float* accum, float* __restrict__ d_out)
{
  int b = blockIdx.x, k = threadIdx.x;
  __shared__ float slp[64], sred[64];
  __shared__ int   skey[64], sorder[64];
  __shared__ float s_m, s_s;
  __shared__ int   s_best;

  int g0 = gt[b * 3 + 0], g1 = gt[b * 3 + 1], g2 = gt[b * 3 + 2];
  int v = (g0 == preds[((size_t)b * 3 + 0) * 64 + k])
        + (g1 == preds[((size_t)b * 3 + 1) * 64 + k])
        + (g2 == preds[((size_t)b * 3 + 2) * 64 + k]);
  skey[k] = v * 64 + (63 - k);            // max key -> max value, first index
  float x = rank_pred[(size_t)b * 64 + k];
  sred[k] = x;
  __syncthreads();
  if (k == 0) {
    float m = sred[0]; int best = skey[0];
    for (int i = 1; i < 64; ++i) { m = fmaxf(m, sred[i]); if (skey[i] > best) best = skey[i]; }
    s_m = m; s_best = best;
  }
  __syncthreads();
  sred[k] = expf(x - s_m);
  __syncthreads();
  if (k == 0) { float s = 0.f; for (int i = 0; i < 64; ++i) s += sred[i]; s_s = s; }
  __syncthreads();
  float lp = x - s_m - logf(s_s);
  slp[k] = lp;
  __syncthreads();
  int r = 0;
  for (int j = 0; j < 64; ++j) {
    float lj = slp[j];
    if (lj > lp || (lj == lp && j < k)) ++r;    // stable descending rank
  }
  sorder[r] = k;
  __syncthreads();
  if (k == 0) {
    int best = s_best, vmax = best >> 6, idx = 63 - (best & 63);
    float mask = (vmax == PP) ? 1.0f : 0.0f;
    atomicAdd(&accum[1], mask);
    atomicAdd(&accum[2], mask * (-slp[idx]));
  }
  int src = sorder[k];
  for (int p = 0; p < 3; ++p)
    d_out[1 + ((size_t)b * 3 + p) * 64 + k] =
        (float)preds[((size_t)b * 3 + p) * 64 + src];
}

__global__ void k_loss_final(const float* accum, float* d_out)
{
  if (threadIdx.x == 0)
    d_out[0] = accum[0] * (1.0f / (float)BB) + accum[2] / fmaxf(accum[1], 1.0f);
}

// ---------------------------------------------------------------------------
// Host launcher
// ---------------------------------------------------------------------------
static inline int cdiv(long a, long b) { return (int)((a + b - 1) / b); }

static inline void launch_gemm(const __bf16* A, int lda, const __bf16* BT,
                               const float* bias, float* Cf, int ldc,
                               __bf16* Cb, int ldo, int M, int N, int Kd,
                               int relu, hipStream_t s)
{
  long waves = (long)(M / 16) * (N / 64);
  gemm_wmma_bf16<<<cdiv(waves, 8), 256, 0, s>>>(A, lda, BT, bias, Cf, ldc, Cb, ldo,
                                                M, N, Kd, relu);
}

extern "C" void kernel_launch(void* const* d_in, const int* in_sizes, int n_in,
                              void* d_out, int out_size, void* d_ws, size_t ws_size,
                              hipStream_t stream)
{
  // ---- inputs (flattened setup_inputs() order) ----
  const float* link_emb = (const float*)d_in[0];   // (8001,64)
  const float* dir_emb  = (const float*)d_in[1];   // (9,32)
  const float* conn_emb = (const float*)d_in[2];
  const float* conn_hyp = (const float*)d_in[3];
  const float* cons_emb = (const float*)d_in[4];
  const float* cons_hyp = (const float*)d_in[5];
  const float* rk_w1 = (const float*)d_in[6];  const float* rk_b1 = (const float*)d_in[7];
  const float* rk_w2 = (const float*)d_in[8];  const float* rk_b2 = (const float*)d_in[9];
  const float* rk_w3 = (const float*)d_in[10]; const float* rk_b3 = (const float*)d_in[11];
  const float* rk_wo = (const float*)d_in[12]; const float* rk_bo = (const float*)d_in[13];
  const float* ml_w1 = (const float*)d_in[14]; const float* ml_b1 = (const float*)d_in[15];
  const float* ml_w2 = (const float*)d_in[16]; const float* ml_b2 = (const float*)d_in[17];
  const float* ml_w3 = (const float*)d_in[18]; const float* ml_b3 = (const float*)d_in[19];
  const float* ml_wo = (const float*)d_in[20]; const float* ml_bo = (const float*)d_in[21];
  const float* tc_w  = (const float*)d_in[22]; const float* tc_b  = (const float*)d_in[23];
  const float* rp_w1 = (const float*)d_in[24]; const float* rp_b1 = (const float*)d_in[25];
  const float* rp_w2 = (const float*)d_in[26]; const float* rp_b2 = (const float*)d_in[27];
  const float* embs  = (const float*)d_in[28];
  const int* preds   = (const int*)d_in[29];
  const int* gt      = (const int*)d_in[30];
  const int* dirlab  = (const int*)d_in[31];
  float* out = (float*)d_out;

  // ---- workspace carve-out (256B aligned regions) ----
  size_t off = 0;
  char* base = (char*)d_ws;
  auto carve = [&](size_t bytes) -> char* {
    char* p = base + off;
    off += (bytes + 255) & ~(size_t)255;
    return p;
  };
  __bf16* featB   = (__bf16*)carve((size_t)BB * FEAT_D * 2);
  __bf16* rank_in = (__bf16*)carve((size_t)BB * RANKIN_D * 2);
  __bf16* mlp_in  = (__bf16*)carve((size_t)BB * MLPIN_D * 2);
  __bf16* hX      = (__bf16*)carve((size_t)BB * HH * 2);
  __bf16* hY      = (__bf16*)carve((size_t)BB * HH * 2);
  __bf16* hZ      = (__bf16*)carve((size_t)BB * HH * 2);
  __bf16* rkW1T   = (__bf16*)carve((size_t)HH * FEAT_D * 2);
  __bf16* rkW2T   = (__bf16*)carve((size_t)HH * HH * 2);
  __bf16* rkW3T   = (__bf16*)carve((size_t)HH * HH * 2);
  __bf16* rkWoT   = (__bf16*)carve((size_t)LOGITS_D * HH * 2);
  __bf16* tcWT    = (__bf16*)carve((size_t)HH * LOGITS_D * 2);
  __bf16* mlW1T   = (__bf16*)carve((size_t)HH * MLPIN_D * 2);
  __bf16* mlW2T   = (__bf16*)carve((size_t)HH * HH * 2);
  __bf16* mlW3T   = (__bf16*)carve((size_t)HH * HH * 2);
  __bf16* mlWoT   = (__bf16*)carve((size_t)PRED_D * HH * 2);
  __bf16* rpW1T   = (__bf16*)carve((size_t)HH * RANKIN_D * 2);
  __bf16* rpW2T   = (__bf16*)carve((size_t)KK * HH * 2);
  float*  Sconn   = (float*)carve((size_t)BB * PP * DE * 4);
  float*  Scons   = (float*)carve((size_t)BB * PP * DE * 4);
  float*  rankprd = (float*)carve((size_t)BB * KK * 4);
  float*  accum   = (float*)carve(256);
  (void)ws_size; (void)n_in; (void)in_sizes; (void)out_size;

  k_zero<<<1, 32, 0, stream>>>(accum);

  // ---- weight conversion + transpose (fp32 -> bf16, W^T) ----
  auto tp = [&](const float* W, __bf16* WT, int Kd, int N) {
    k_transpose<<<cdiv((long)Kd * N, 256), 256, 0, stream>>>(W, WT, Kd, N);
  };
  tp(rk_w1, rkW1T, FEAT_D, HH);     tp(rk_w2, rkW2T, HH, HH);
  tp(rk_w3, rkW3T, HH, HH);         tp(rk_wo, rkWoT, HH, LOGITS_D);
  tp(tc_w,  tcWT,  LOGITS_D, HH);
  tp(ml_w1, mlW1T, MLPIN_D, HH);    tp(ml_w2, mlW2T, HH, HH);
  tp(ml_w3, mlW3T, HH, HH);         tp(ml_wo, mlWoT, HH, PRED_D);
  tp(rp_w1, rpW1T, RANKIN_D, HH);   tp(rp_w2, rpW2T, HH, KK);

  // ---- gathers / elementwise features ----
  k_feat<<<cdiv((long)BB * FEAT_D, 256), 256, 0, stream>>>(preds, dirlab, link_emb,
                                                           dir_emb, featB);
  k_sums<<<cdiv((long)BB * PP * DE, 256), 256, 0, stream>>>(preds, link_emb, conn_hyp,
                                                            cons_hyp, Sconn, Scons);
  k_connm<<<cdiv((long)BB * KK * DE, 256), 256, 0, stream>>>(preds, link_emb, conn_hyp,
                                                             cons_hyp, conn_emb, cons_emb,
                                                             Sconn, Scons, rank_in);
  k_embs<<<cdiv((long)BB * 128, 256), 256, 0, stream>>>(embs, mlp_in);

  // ---- rank MLP -> logits (bf16 into rank_in[:, :16384]) ----
  launch_gemm(featB, FEAT_D, rkW1T, rk_b1, nullptr, 0, hX, HH, BB, HH, FEAT_D, 1, stream);
  launch_gemm(hX, HH, rkW2T, rk_b2, nullptr, 0, hY, HH, BB, HH, HH, 1, stream);
  launch_gemm(hY, HH, rkW3T, rk_b3, nullptr, 0, hZ, HH, BB, HH, HH, 0, stream);
  launch_gemm(hZ, HH, rkWoT, rk_bo, nullptr, 0, rank_in, RANKIN_D, BB, LOGITS_D, HH, 0, stream);

  // ---- top1_logits = logits @ tc_w + tc_b  (bf16 into mlp_in[:, 128:]) ----
  launch_gemm(rank_in, RANKIN_D, tcWT, tc_b, nullptr, 0, mlp_in + 128, MLPIN_D,
              BB, HH, LOGITS_D, 0, stream);

  // ---- mlp -> top1_pred (fp32 straight into d_out) ----
  launch_gemm(mlp_in, MLPIN_D, mlW1T, ml_b1, nullptr, 0, hX, HH, BB, HH, MLPIN_D, 1, stream);
  launch_gemm(hX, HH, mlW2T, ml_b2, nullptr, 0, hY, HH, BB, HH, HH, 1, stream);
  launch_gemm(hY, HH, mlW3T, ml_b3, nullptr, 0, hZ, HH, BB, HH, HH, 0, stream);
  launch_gemm(hZ, HH, mlWoT, ml_bo, out + OUT_PRED_OFF, PRED_D, nullptr, 0,
              BB, PRED_D, HH, 0, stream);

  // ---- rank-order head ----
  launch_gemm(rank_in, RANKIN_D, rpW1T, rp_b1, nullptr, 0, hX, HH, BB, HH, RANKIN_D, 1, stream);
  launch_gemm(hX, HH, rpW2T, rp_b2, rankprd, KK, nullptr, 0, BB, KK, HH, 0, stream);

  // ---- losses + sorted preds ----
  k_cls_loss<<<BB * PP, 256, 0, stream>>>(out + OUT_PRED_OFF, gt, accum);
  k_final<<<BB, 64, 0, stream>>>(rankprd, preds, gt, accum, out);
  k_loss_final<<<1, 32, 0, stream>>>(accum, out);
}
